// FreqEnhancedAttention_22247930593569
// MI455X (gfx1250) — matrix operand order repaired and checked by reading, not compiled, not run on hardware
//
#include <hip/hip_runtime.h>
#include <hip/hip_bf16.h>
#include <stdint.h>

// ---------------------------------------------------------------------------
// FreqEnhancedAttention for MI455X (gfx1250, wave32, WMMA bf16 16x16x32)
// B=2, L=S=2048, D_MODEL=1024, H=16, dk=64
// Register-blocked strips per wave + ping-pong (unroll-by-2) k-loops:
// no loop-carried buffer copies, loads overlap WMMAs with partial waits.
// ---------------------------------------------------------------------------

#define DMODEL 1024
#define NH     16
#define DK     64
#define BSZ    2
#define LQ     2048
#define SSZ    2048
#define MROWS  (BSZ * LQ)        // 4096 flattened token rows

typedef __attribute__((ext_vector_type(16))) __bf16 v16bf;
typedef __attribute__((ext_vector_type(8)))  float  v8f;

union Frag16 { uint4 q[2]; v16bf v; };

__device__ __forceinline__ uint16_t f2bf(float f) {
    // round-to-nearest-even f32 -> bf16
    unsigned u = __float_as_uint(f);
    u += 0x7fffu + ((u >> 16) & 1u);
    return (uint16_t)(u >> 16);
}

__device__ __forceinline__ v8f wmma_bf16(v16bf a, v16bf b, v8f c) {
    // D = A(16x32 bf16) * B(32x16 bf16) + C(16x16 f32)
    return __builtin_amdgcn_wmma_f32_16x16x32_bf16(
        /*neg_a=*/false, a, /*neg_b=*/false, b,
        /*c_mod=*/(short)0, c, /*reuse_a=*/false, /*reuse_b=*/false);
}

// A fragment: 16x32 bf16 tile, row-major source with leading dim `lda` (elems).
// ISA layout: lanes 0-15 hold rows M=0..15 with K={0..7,16..23};
//             lanes 16-31 hold same rows with K={8..15,24..31}.
__device__ __forceinline__ v16bf load_a(const uint16_t* __restrict__ A,
                                        int lda, int m0, int k0) {
    int lane = threadIdx.x & 31;
    const uint16_t* p = A + (size_t)(m0 + (lane & 15)) * lda
                          + k0 + ((lane >> 4) << 3);
    Frag16 f;
    f.q[0] = *(const uint4*)(p);        // K .. K+7
    f.q[1] = *(const uint4*)(p + 16);   // K+16 .. K+23
    return f.v;
}

// B fragment: 32x16 bf16 tile; source is B^T row-major ([N][K], leading dim
// `ldb` elems); same K layout as the A fragment.
// Lanes 0-15 cover K={0..7,16..23}, lanes 16-31 cover K={8..15,24..31}.
__device__ __forceinline__ v16bf load_b(const uint16_t* __restrict__ Bt,
                                        int ldb, int n0, int k0) {
    int lane = threadIdx.x & 31;
    const uint16_t* p = Bt + (size_t)(n0 + (lane & 15)) * ldb
                           + k0 + ((lane >> 4) << 3);
    Frag16 f;
    f.q[0] = *(const uint4*)(p);        // K .. K+7
    f.q[1] = *(const uint4*)(p + 16);   // K+16 .. K+23
    return f.v;
}

// ---------------------------------------------------------------------------
// f32 -> bf16 elementwise convert (activations)
// ---------------------------------------------------------------------------
__global__ __launch_bounds__(256)
void cvt_bf16_kernel(const float* __restrict__ in, uint16_t* __restrict__ out,
                     long n) {
    long i = (long)blockIdx.x * blockDim.x + threadIdx.x;
    long stride = (long)gridDim.x * blockDim.x;
    for (; i < n; i += stride) out[i] = f2bf(in[i]);
}

// ---------------------------------------------------------------------------
// Weight convert + transpose: Wt[n][k] = bf16(W[k][n]), 1024x1024,
// LDS-tiled 32x32 so both global read and write are coalesced.
// ---------------------------------------------------------------------------
__global__ __launch_bounds__(256)
void cvt_wT_kernel(const float* __restrict__ W, uint16_t* __restrict__ Wt) {
    __shared__ uint16_t tile[32][33];            // +1 pad: no bank conflicts
    int bx = blockIdx.x & 31;                    // n-tile
    int by = blockIdx.x >> 5;                    // k-tile
    int tx = threadIdx.x & 31;
    int ty = threadIdx.x >> 5;                   // 0..7
#pragma unroll
    for (int i = 0; i < 4; ++i) {
        int k = by * 32 + ty + i * 8;
        int n = bx * 32 + tx;
        tile[ty + i * 8][tx] = f2bf(W[(size_t)k * DMODEL + n]);
    }
    __syncthreads();
#pragma unroll
    for (int i = 0; i < 4; ++i) {
        int n = bx * 32 + ty + i * 8;
        int k = by * 32 + tx;
        Wt[(size_t)n * DMODEL + k] = tile[tx][ty + i * 8];
    }
}

// ---------------------------------------------------------------------------
// Dual-GEMM projection, 16x32 per wave per GEMM (NB=2), ping-pong k-loop:
//   acc1[j] = X1 @ W1^T ; acc2[j] = X2 @ W2^T    (K = DMODEL, bf16 WMMA)
//   gate==0 : y = (acc1+b1) + (acc2+b2)           (Q, V paths)
//   gate==1 : y = (acc1+b1) * sigmoid(acc2+b2)    (K path)
//   headT==0: store Y[m][n] bf16 (natural [tokens][channels])
//   headT==1: store Y[b][h][d][s] bf16 (head-transposed, for V as GEMM-B)
// ---------------------------------------------------------------------------
__global__ __launch_bounds__(256)
void proj_dual_kernel(const uint16_t* __restrict__ X1,
                      const uint16_t* __restrict__ W1t,
                      const float*    __restrict__ b1,
                      const uint16_t* __restrict__ X2,
                      const uint16_t* __restrict__ W2t,
                      const float*    __restrict__ b2,
                      uint16_t* __restrict__ Y, int gate, int headT) {
    int wave = threadIdx.x >> 5;
    int tile = blockIdx.x * 8 + wave;            // over (M/16)*(DMODEL/32)
    int nt = DMODEL / 32;                        // 32 n-blocks of width 32
    int m0 = (tile / nt) * 16;
    int n0 = (tile % nt) * 32;

    v8f acc1[2] = {}; v8f acc2[2] = {};

    // ping buffer: step k = 0
    v16bf a1A = load_a(X1, DMODEL, m0, 0);
    v16bf a2A = load_a(X2, DMODEL, m0, 0);
    v16bf w1A[2], w2A[2];
#pragma unroll
    for (int j = 0; j < 2; ++j) {
        w1A[j] = load_b(W1t, DMODEL, n0 + 16 * j, 0);
        w2A[j] = load_b(W2t, DMODEL, n0 + 16 * j, 0);
    }

    // iteration at k0 consumes steps (k0-32) and (k0), preloads (k0+32).
    for (int k0 = 32; k0 < DMODEL - 32; k0 += 64) {
        v16bf a1B = load_a(X1, DMODEL, m0, k0);
        v16bf a2B = load_a(X2, DMODEL, m0, k0);
        v16bf w1B[2], w2B[2];
#pragma unroll
        for (int j = 0; j < 2; ++j) {
            w1B[j] = load_b(W1t, DMODEL, n0 + 16 * j, k0);
            w2B[j] = load_b(W2t, DMODEL, n0 + 16 * j, k0);
        }
#pragma unroll
        for (int j = 0; j < 2; ++j) acc1[j] = wmma_bf16(a1A, w1A[j], acc1[j]);
#pragma unroll
        for (int j = 0; j < 2; ++j) acc2[j] = wmma_bf16(a2A, w2A[j], acc2[j]);

        a1A = load_a(X1, DMODEL, m0, k0 + 32);
        a2A = load_a(X2, DMODEL, m0, k0 + 32);
#pragma unroll
        for (int j = 0; j < 2; ++j) {
            w1A[j] = load_b(W1t, DMODEL, n0 + 16 * j, k0 + 32);
            w2A[j] = load_b(W2t, DMODEL, n0 + 16 * j, k0 + 32);
        }
#pragma unroll
        for (int j = 0; j < 2; ++j) acc1[j] = wmma_bf16(a1B, w1B[j], acc1[j]);
#pragma unroll
        for (int j = 0; j < 2; ++j) acc2[j] = wmma_bf16(a2B, w2B[j], acc2[j]);
    }
    // tail: ping buffer holds k-step DMODEL-64; step DMODEL-32 is never loaded
#pragma unroll
    for (int j = 0; j < 2; ++j) acc1[j] = wmma_bf16(a1A, w1A[j], acc1[j]);
#pragma unroll
    for (int j = 0; j < 2; ++j) acc2[j] = wmma_bf16(a2A, w2A[j], acc2[j]);

    int lane = threadIdx.x & 31;
    int cm = m0 + ((lane >> 4) << 3);
#pragma unroll
    for (int j = 0; j < 2; ++j) {
        int cn = n0 + 16 * j + (lane & 15);
        float bias1 = b1[cn], bias2 = b2[cn];
#pragma unroll
        for (int v = 0; v < 8; ++v) {
            float x1 = acc1[j][v] + bias1;
            float x2 = acc2[j][v] + bias2;
            float y  = gate ? x1 * (1.0f / (1.0f + __expf(-x2))) : (x1 + x2);
            int m = cm + v;
            if (!headT) {
                Y[(size_t)m * DMODEL + cn] = f2bf(y);
            } else {
                int b = m / SSZ, s = m % SSZ;
                int h = cn / DK, d = cn % DK;
                Y[(((size_t)b * NH + h) * DK + d) * SSZ + s] = f2bf(y);
            }
        }
    }
}

// ---------------------------------------------------------------------------
// scores[b,h,l,s] = (Q_h @ K_h^T) / 8, masked with -1e9 where attn_mask!=0.
// 16(l) x 64(s) per wave (NB=4); K = dk = 64 -> only 2 k-steps, fully
// unrolled (compiler hoists all loads). Writes f32 into attn region of d_out.
// ---------------------------------------------------------------------------
__global__ __launch_bounds__(256)
void scores_kernel(const uint16_t* __restrict__ Qb,
                   const uint16_t* __restrict__ Kb,
                   const unsigned char* __restrict__ mask,
                   float* __restrict__ attn) {
    int wave = threadIdx.x >> 5;
    long tile = (long)blockIdx.x * 8 + wave;     // B*H*(L/16)*(S/64) tiles
    int s0 = (int)(tile % (SSZ / 64)) * 64; tile /= (SSZ / 64);
    int l0 = (int)(tile % (LQ / 16)) * 16;  tile /= (LQ / 16);
    int h  = (int)(tile % NH);              tile /= NH;
    int b  = (int)tile;

    const uint16_t* Qh = Qb + (size_t)b * LQ  * DMODEL + h * DK;
    const uint16_t* Kh = Kb + (size_t)b * SSZ * DMODEL + h * DK;

    v8f acc[4] = {};
#pragma unroll
    for (int k0 = 0; k0 < DK; k0 += 32) {
        v16bf a = load_a(Qh, DMODEL, l0, k0);
#pragma unroll
        for (int j = 0; j < 4; ++j)
            acc[j] = wmma_bf16(a, load_b(Kh, DMODEL, s0 + 16 * j, k0), acc[j]);
    }

    int lane = threadIdx.x & 31;
    int lm = l0 + ((lane >> 4) << 3);
    const unsigned char* mrow = mask + (size_t)b * LQ * SSZ;
    float* arow = attn + ((size_t)b * NH + h) * LQ * SSZ;
#pragma unroll
    for (int j = 0; j < 4; ++j) {
        int sn = s0 + 16 * j + (lane & 15);
#pragma unroll
        for (int v = 0; v < 8; ++v) {
            int l = lm + v;
            float sc = acc[j][v] * 0.125f;       // 1/sqrt(64)
            if (mrow[(size_t)l * SSZ + sn]) sc = -1e9f;
            arow[(size_t)l * SSZ + sn] = sc;
        }
    }
}

// ---------------------------------------------------------------------------
// Row softmax over S=2048, in place in d_out (f32), plus fused bf16 copy for
// the P@V WMMA pass. One 256-thread block per row, 8 elems/thread.
// ---------------------------------------------------------------------------
__global__ __launch_bounds__(256)
void softmax_kernel(float* __restrict__ attn, uint16_t* __restrict__ Pb) {
    long row = blockIdx.x;
    float*    p  = attn + row * (long)SSZ;
    uint16_t* pb = Pb   + row * (long)SSZ;
    __shared__ float red[256];
    int t = threadIdx.x;

    float vals[8];
    float vmax = -3.0e38f;
#pragma unroll
    for (int i = 0; i < 8; ++i) {
        vals[i] = p[t + i * 256];
        vmax = fmaxf(vmax, vals[i]);
    }
    red[t] = vmax; __syncthreads();
    for (int off = 128; off > 0; off >>= 1) {
        if (t < off) red[t] = fmaxf(red[t], red[t + off]);
        __syncthreads();
    }
    vmax = red[0]; __syncthreads();

    float sum = 0.0f;
#pragma unroll
    for (int i = 0; i < 8; ++i) { vals[i] = __expf(vals[i] - vmax); sum += vals[i]; }
    red[t] = sum; __syncthreads();
    for (int off = 128; off > 0; off >>= 1) {
        if (t < off) red[t] += red[t + off];
        __syncthreads();
    }
    float inv = 1.0f / red[0];
#pragma unroll
    for (int i = 0; i < 8; ++i) {
        float w = vals[i] * inv;
        p[t + i * 256]  = w;
        pb[t + i * 256] = f2bf(w);
    }
}

// ---------------------------------------------------------------------------
// AO[b,l,h*64+d] = sum_s P[b,h,l,s] * V[b,s,h,d]   (P bf16, Vt = [b][h][d][s])
// One wave produces a full 16(l) x 64(d) strip; 64-step ping-pong k-loop
// over S so P/V loads overlap WMMAs with no register copies.
// ---------------------------------------------------------------------------
__global__ __launch_bounds__(256)
void attnv_kernel(const uint16_t* __restrict__ Pb,
                  const uint16_t* __restrict__ Vt,
                  uint16_t* __restrict__ AO) {
    int wave = threadIdx.x >> 5;
    long tile = (long)blockIdx.x * 8 + wave;     // B*H*(L/16) tiles
    int l0 = (int)(tile % (LQ / 16)) * 16; tile /= (LQ / 16);
    int h  = (int)(tile % NH);             tile /= NH;
    int b  = (int)tile;

    const uint16_t* Ph = Pb + ((size_t)b * NH + h) * LQ * SSZ;
    const uint16_t* Vh = Vt + ((size_t)b * NH + h) * DK * SSZ;

    v8f acc[4] = {};

    v16bf aA = load_a(Ph, SSZ, l0, 0);
    v16bf bA[4];
#pragma unroll
    for (int j = 0; j < 4; ++j) bA[j] = load_b(Vh, SSZ, 16 * j, 0);

    for (int k0 = 32; k0 < SSZ - 32; k0 += 64) {
        v16bf aB = load_a(Ph, SSZ, l0, k0);
        v16bf bB[4];
#pragma unroll
        for (int j = 0; j < 4; ++j) bB[j] = load_b(Vh, SSZ, 16 * j, k0);
#pragma unroll
        for (int j = 0; j < 4; ++j) acc[j] = wmma_bf16(aA, bA[j], acc[j]);

        aA = load_a(Ph, SSZ, l0, k0 + 32);
#pragma unroll
        for (int j = 0; j < 4; ++j) bA[j] = load_b(Vh, SSZ, 16 * j, k0 + 32);
#pragma unroll
        for (int j = 0; j < 4; ++j) acc[j] = wmma_bf16(aB, bB[j], acc[j]);
    }
#pragma unroll
    for (int j = 0; j < 4; ++j) acc[j] = wmma_bf16(aA, bA[j], acc[j]);

    int lane = threadIdx.x & 31;
    int lm = l0 + ((lane >> 4) << 3);
#pragma unroll
    for (int j = 0; j < 4; ++j) {
        int dn = 16 * j + (lane & 15);
#pragma unroll
        for (int v = 0; v < 8; ++v) {
            int l = lm + v;
            AO[((size_t)b * LQ + l) * DMODEL + h * DK + dn] = f2bf(acc[j][v]);
        }
    }
}

// ---------------------------------------------------------------------------
// Final projection: Out = AO @ Wo + bo, f32 output (first region of d_out).
// 16x64 per wave (NB=4), 32-step ping-pong k-loop.
// ---------------------------------------------------------------------------
__global__ __launch_bounds__(256)
void outproj_kernel(const uint16_t* __restrict__ A,
                    const uint16_t* __restrict__ Wt,
                    const float* __restrict__ bias,
                    float* __restrict__ Out) {
    int wave = threadIdx.x >> 5;
    int tile = blockIdx.x * 8 + wave;            // over (M/16)*(DMODEL/64)
    int nt = DMODEL / 64;                        // 16
    int m0 = (tile / nt) * 16;
    int n0 = (tile % nt) * 64;

    v8f acc[4] = {};

    v16bf aA = load_a(A, DMODEL, m0, 0);
    v16bf bA[4];
#pragma unroll
    for (int j = 0; j < 4; ++j) bA[j] = load_b(Wt, DMODEL, n0 + 16 * j, 0);

    for (int k0 = 32; k0 < DMODEL - 32; k0 += 64) {
        v16bf aB = load_a(A, DMODEL, m0, k0);
        v16bf bB[4];
#pragma unroll
        for (int j = 0; j < 4; ++j) bB[j] = load_b(Wt, DMODEL, n0 + 16 * j, k0);
#pragma unroll
        for (int j = 0; j < 4; ++j) acc[j] = wmma_bf16(aA, bA[j], acc[j]);

        aA = load_a(A, DMODEL, m0, k0 + 32);
#pragma unroll
        for (int j = 0; j < 4; ++j) bA[j] = load_b(Wt, DMODEL, n0 + 16 * j, k0 + 32);
#pragma unroll
        for (int j = 0; j < 4; ++j) acc[j] = wmma_bf16(aB, bB[j], acc[j]);
    }
#pragma unroll
    for (int j = 0; j < 4; ++j) acc[j] = wmma_bf16(aA, bA[j], acc[j]);

    int lane = threadIdx.x & 31;
    int cm = m0 + ((lane >> 4) << 3);
#pragma unroll
    for (int j = 0; j < 4; ++j) {
        int cn = n0 + 16 * j + (lane & 15);
        float bv = bias[cn];
#pragma unroll
        for (int v = 0; v < 8; ++v)
            Out[(size_t)(cm + v) * DMODEL + cn] = acc[j][v] + bv;
    }
}

// ---------------------------------------------------------------------------
// Host-side orchestration
// ---------------------------------------------------------------------------
extern "C" void kernel_launch(void* const* d_in, const int* in_sizes, int n_in,
                              void* d_out, int out_size, void* d_ws, size_t ws_size,
                              hipStream_t stream) {
    const float* queries   = (const float*)d_in[0];
    const float* keys      = (const float*)d_in[1];
    const float* values    = (const float*)d_in[2];
    const unsigned char* attn_mask = (const unsigned char*)d_in[3]; // bool [B,1,L,S]
    const float* context   = (const float*)d_in[4];
    const float* Wq  = (const float*)d_in[5],  *bq  = (const float*)d_in[6];
    const float* Wk  = (const float*)d_in[7],  *bk  = (const float*)d_in[8];
    const float* Wv  = (const float*)d_in[9],  *bv  = (const float*)d_in[10];
    const float* Wqb = (const float*)d_in[11], *bqb = (const float*)d_in[12];
    const float* Wkg = (const float*)d_in[13], *bkg = (const float*)d_in[14];
    const float* Wvb = (const float*)d_in[15], *bvb = (const float*)d_in[16];
    const float* Wo  = (const float*)d_in[17], *bo  = (const float*)d_in[18];

    // d_out layout: out [B,L,DMODEL] f32, then attn_weights [B,H,L,S] f32
    float* out_proj = (float*)d_out;
    float* attn     = out_proj + (size_t)MROWS * DMODEL;

    // Workspace carving (all 256B aligned)
    size_t off = 0;
    auto carve = [&](size_t bytes) {
        void* p = (char*)d_ws + off;
        off += (bytes + 255) & ~(size_t)255;
        return p;
    };
    const size_t ACT = (size_t)MROWS * DMODEL * sizeof(uint16_t);   // 8 MiB
    const size_t WT  = (size_t)DMODEL * DMODEL * sizeof(uint16_t);  // 2 MiB
    uint16_t* Xq  = (uint16_t*)carve(ACT);
    uint16_t* Xk  = (uint16_t*)carve(ACT);
    uint16_t* Xv  = (uint16_t*)carve(ACT);
    uint16_t* Xc  = (uint16_t*)carve(ACT);
    uint16_t* WqT  = (uint16_t*)carve(WT);
    uint16_t* WkT  = (uint16_t*)carve(WT);
    uint16_t* WvT  = (uint16_t*)carve(WT);
    uint16_t* WqbT = (uint16_t*)carve(WT);
    uint16_t* WkgT = (uint16_t*)carve(WT);
    uint16_t* WvbT = (uint16_t*)carve(WT);
    uint16_t* WoT  = (uint16_t*)carve(WT);
    uint16_t* Qbf = (uint16_t*)carve(ACT);
    uint16_t* Kbf = (uint16_t*)carve(ACT);
    uint16_t* Vt  = (uint16_t*)carve(ACT);              // [b][h][d][s]
    uint16_t* AO  = (uint16_t*)carve(ACT);
    uint16_t* Pbf = (uint16_t*)carve((size_t)BSZ * NH * LQ * SSZ * sizeof(uint16_t)); // 256 MiB

    const long nact = (long)MROWS * DMODEL;

    // 1) activations f32 -> bf16
    cvt_bf16_kernel<<<2048, 256, 0, stream>>>(queries, Xq, nact);
    cvt_bf16_kernel<<<2048, 256, 0, stream>>>(keys,    Xk, nact);
    cvt_bf16_kernel<<<2048, 256, 0, stream>>>(values,  Xv, nact);
    cvt_bf16_kernel<<<2048, 256, 0, stream>>>(context, Xc, nact);

    // 2) weights f32 -> bf16, transposed to [N][K] for contiguous B-fragments
    cvt_wT_kernel<<<1024, 256, 0, stream>>>(Wq,  WqT);
    cvt_wT_kernel<<<1024, 256, 0, stream>>>(Wk,  WkT);
    cvt_wT_kernel<<<1024, 256, 0, stream>>>(Wv,  WvT);
    cvt_wT_kernel<<<1024, 256, 0, stream>>>(Wqb, WqbT);
    cvt_wT_kernel<<<1024, 256, 0, stream>>>(Wkg, WkgT);
    cvt_wT_kernel<<<1024, 256, 0, stream>>>(Wvb, WvbT);
    cvt_wT_kernel<<<1024, 256, 0, stream>>>(Wo,  WoT);

    // 3) fused dual projections: (M/16)*(D/32)=8192 strips, 8 waves/block
    const int projBlocks = (MROWS / 16) * (DMODEL / 32) / 8;   // 1024
    // Q = queries@Wq + bq + context@Wqb + bqb
    proj_dual_kernel<<<projBlocks, 256, 0, stream>>>(Xq, WqT, bq, Xc, WqbT, bqb,
                                                     Qbf, /*gate=*/0, /*headT=*/0);
    // K = (keys@Wk + bk) * sigmoid(context@Wkg + bkg)
    proj_dual_kernel<<<projBlocks, 256, 0, stream>>>(Xk, WkT, bk, Xc, WkgT, bkg,
                                                     Kbf, /*gate=*/1, /*headT=*/0);
    // V = values@Wv + bv + context@Wvb + bvb   (stored head-transposed)
    proj_dual_kernel<<<projBlocks, 256, 0, stream>>>(Xv, WvT, bv, Xc, WvbT, bvb,
                                                     Vt, /*gate=*/0, /*headT=*/1);

    // 4) scores (+scale, +mask) -> f32 directly into attn region of d_out
    const int scoreBlocks = BSZ * NH * (LQ / 16) * (SSZ / 64) / 8; // 16384
    scores_kernel<<<scoreBlocks, 256, 0, stream>>>(Qbf, Kbf, attn_mask, attn);

    // 5) in-place softmax + fused bf16 copy of P
    softmax_kernel<<<BSZ * NH * LQ, 256, 0, stream>>>(attn, Pbf);

    // 6) P @ V per head -> AO bf16 (one wave = full 16x64 strip)
    const int avBlocks = BSZ * NH * (LQ / 16) / 8;                 // 512
    attnv_kernel<<<avBlocks, 256, 0, stream>>>(Pbf, Vt, AO);

    // 7) output projection -> f32 d_out
    const int outBlocks = (MROWS / 16) * (DMODEL / 64) / 8;        // 512
    outproj_kernel<<<outBlocks, 256, 0, stream>>>(AO, WoT, bo, out_proj);

    (void)in_sizes; (void)n_in; (void)out_size; (void)ws_size;
}
